// CoPE_2d_unit_1769526526340
// MI455X (gfx1250) — matrix operand order repaired and checked by reading, not compiled, not run on hardware
//
#include <hip/hip_runtime.h>
#include <hip/hip_bf16.h>

// CoPE kernel for MI455X (gfx1250, wave32, WMMA).
// out[b,q,k] = lerp(logits_int[b,q, floor(pos)], logits_int[b,q, ceil(pos)], frac(pos))
// pos[b,q,k] = min(revcumsum_k sigmoid((q*SCALE)·K), 63)

#define SCALE   0.125f
#define S_DIM   1024
#define D_DIM   64
#define NPOS    64
#define QT      16      // query rows per workgroup
#define ROWSTR  1060    // padded LDS row stride (words) for the pos/logits buffer
// Insert one pad word every 32 elements so lane-stride-32 accesses in the scan
// phase map to 32 distinct banks (addr = 33*lane + j mod 64, 33 odd -> bijective).
#define LIDX(r, kk) ((r) * ROWSTR + (kk) + ((kk) >> 5))

typedef __attribute__((ext_vector_type(16))) __bf16 v16bf;
typedef __attribute__((ext_vector_type(8)))  float  v8f;

// Load a 16x32 bf16 A-fragment slice from a row-major f32 tile (row stride D_DIM).
// A layout (matches B): lane<16 -> M=lane, K = 0..15;
// lane>=16 -> M=lane-16, K = 16..31; elements packed 2/VGPR in order.
__device__ __forceinline__ void load_a_frag(const float* __restrict__ tile,
                                            int m, int koff, int kbase,
                                            v16bf& a) {
  const float* row = tile + m * D_DIM + kbase;
#pragma unroll
  for (int i = 0; i < 8; ++i) {
    a[i]     = (__bf16)row[koff + i];
    a[8 + i] = (__bf16)row[koff + 8 + i];
  }
}

__global__ __launch_bounds__(512) void cope_kernel(
    const float* __restrict__ q, const float* __restrict__ kmat,
    const float* __restrict__ pos_emb, const int* __restrict__ is_cope_k,
    float* __restrict__ out) {
  extern __shared__ float smem[];
  float* s_pos = smem;                   // [QT][ROWSTR] logits -> positions
  float* s_li  = smem + QT * ROWSTR;     // [QT][NPOS]   logits_int tile

  const int lane  = threadIdx.x & 31;
  const int wave  = threadIdx.x >> 5;    // 0..15
  const int b     = blockIdx.x >> 6;     // S_DIM/QT == 64 tiles per batch
  const int qbase = (blockIdx.x & 63) * QT;

  const float* qtile = q    + ((size_t)b * S_DIM + qbase) * D_DIM;
  const float* kbat  = kmat + (size_t)b * S_DIM * D_DIM;
  const float* btile = (is_cope_k[0] != 0)
                         ? qtile
                         : (kmat + ((size_t)b * S_DIM + qbase) * D_DIM);

  // ---------------- A fragments (Q tile 16x64, unscaled bf16) ----------------
  const int am   = lane & 15;
  const int koff = (lane >> 4) * 16;
  v16bf aq0, aq1, ab0, ab1;
  load_a_frag(qtile, am, koff, 0,  aq0);
  load_a_frag(qtile, am, koff, 32, aq1);
  load_a_frag(btile, am, koff, 0,  ab0);   // base for logits_int (q when is_cope_k)
  load_a_frag(btile, am, koff, 32, ab1);

  // B-fragment lane mapping (K x 16 bf16): lane<16 -> N=lane, K=0..15 packed;
  // lane>=16 -> N=lane-16, K=16..31 packed (mirrors documented sparse-B layout).
  const int bn    = lane & 15;
  const int dbase = (lane >> 4) * 16;
  const int mbase = (lane >> 4) * 8;     // C layout: VGPR r -> M = mbase + r

  // ---------------- Phase 1: attn logits, wave w -> keys [64w, 64w+64) -------
#pragma unroll
  for (int tt = 0; tt < 4; ++tt) {
    const int t0 = wave * 64 + tt * 16;
    const float* krow = kbat + (size_t)(t0 + bn) * D_DIM;   // 64B contiguous/lane
    v16bf b0, b1;
#pragma unroll
    for (int i = 0; i < 16; ++i) {
      b0[i] = (__bf16)(SCALE * krow[dbase + i]);            // K dims 0..31
      b1[i] = (__bf16)(SCALE * krow[32 + dbase + i]);       // K dims 32..63
    }
    v8f c = {0.f, 0.f, 0.f, 0.f, 0.f, 0.f, 0.f, 0.f};
    c = __builtin_amdgcn_wmma_f32_16x16x32_bf16(false, aq0, false, b0,
                                                (short)0, c, false, false);
    c = __builtin_amdgcn_wmma_f32_16x16x32_bf16(false, aq1, false, b1,
                                                (short)0, c, false, false);
#pragma unroll
    for (int r = 0; r < 8; ++r)
      s_pos[LIDX(mbase + r, t0 + bn)] = c[r];
  }

  // ---------------- Phase 1b: logits_int = base x pos_emb (waves 0..3) -------
  if (wave < 4) {                         // wave-uniform: EXEC stays all-ones
    const int n0 = wave * 16;
    v16bf p0, p1;
#pragma unroll
    for (int i = 0; i < 16; ++i) {        // pos_emb is [D][NPOS] row-major
      p0[i] = (__bf16)pos_emb[(size_t)(dbase + i) * NPOS + n0 + bn];
      p1[i] = (__bf16)pos_emb[(size_t)(32 + dbase + i) * NPOS + n0 + bn];
    }
    v8f c = {0.f, 0.f, 0.f, 0.f, 0.f, 0.f, 0.f, 0.f};
    c = __builtin_amdgcn_wmma_f32_16x16x32_bf16(false, ab0, false, p0,
                                                (short)0, c, false, false);
    c = __builtin_amdgcn_wmma_f32_16x16x32_bf16(false, ab1, false, p1,
                                                (short)0, c, false, false);
#pragma unroll
    for (int r = 0; r < 8; ++r)
      s_li[(mbase + r) * NPOS + n0 + bn] = c[r];
  }

  __syncthreads();

  // ---------------- Phase 2: sigmoid + reverse cumsum; wave w owns row w -----
  {
    float suf[32];
    float run = 0.f;
#pragma unroll
    for (int j = 31; j >= 0; --j) {       // lane owns k in [32*lane, 32*lane+32)
      float x = s_pos[LIDX(wave, lane * 32 + j)];   // conflict-free (33*lane+j)
      float g = 1.0f / (1.0f + __expf(-x));
      run += g;
      suf[j] = run;                       // local reverse suffix sum
    }
    // Kogge-Stone inclusive *suffix* scan of per-lane totals across the wave.
    float v = run;
#pragma unroll
    for (int d = 1; d < 32; d <<= 1) {
      float t = __shfl_down(v, d, 32);
      if (lane + d < 32) v += t;
    }
    const float off = v - run;            // exclusive suffix (lanes > lane)
#pragma unroll
    for (int j = 0; j < 32; ++j) {
      float p = fminf(suf[j] + off, (float)(NPOS - 1));
      s_pos[LIDX(wave, lane * 32 + j)] = p;
    }
  }

  __syncthreads();

  // ---------------- Phase 3: gather + lerp, coalesced float4 stores ----------
  float* orow = out + ((size_t)b * S_DIM + qbase + wave) * S_DIM;
  const float* lirow = s_li + wave * NPOS;
#pragma unroll
  for (int j = 0; j < 8; ++j) {
    const int k0 = j * 128 + lane * 4;    // wave writes 512 contiguous bytes/iter
    float vals[4];
#pragma unroll
    for (int cc = 0; cc < 4; ++cc) {
      float p  = s_pos[LIDX(wave, k0 + cc)];
      float pf = floorf(p);
      int   fi = (int)pf;
      int   ci = (int)ceilf(p);
      fi = max(0, min(NPOS - 1, fi));
      ci = max(0, min(NPOS - 1, ci));
      float wt = p - pf;
      vals[cc] = lirow[ci] * wt + lirow[fi] * (1.0f - wt);
    }
    float4 o = make_float4(vals[0], vals[1], vals[2], vals[3]);
    *reinterpret_cast<float4*>(orow + k0) = o;   // 16B aligned: k0 % 4 == 0
  }
}

extern "C" void kernel_launch(void* const* d_in, const int* in_sizes, int n_in,
                              void* d_out, int out_size, void* d_ws, size_t ws_size,
                              hipStream_t stream) {
  (void)n_in; (void)out_size; (void)d_ws; (void)ws_size;
  const float* q       = (const float*)d_in[0];
  const float* k       = (const float*)d_in[1];
  // d_in[2] = v (unused by the reference output), d_in[4] = w_k (unused)
  const float* pos_emb = (const float*)d_in[3];
  const int*   is_cope = (const int*)d_in[5];
  float*       out     = (float*)d_out;

  const int B = in_sizes[0] / (S_DIM * D_DIM);   // 32
  dim3 grid(B * (S_DIM / QT));                   // 2048 workgroups
  dim3 block(512);                               // 16 waves (wave32)
  size_t lds_bytes = (size_t)(QT * ROWSTR + QT * NPOS) * sizeof(float); // ~72 KB
  hipLaunchKernelGGL(cope_kernel, grid, block, lds_bytes, stream,
                     q, k, pos_emb, is_cope, out);
}
